// HandInfer_44598940401960
// MI455X (gfx1250) — matrix operand order repaired and checked by reading, not compiled, not run on hardware
//
#include <hip/hip_runtime.h>
#include <hip/hip_fp16.h>
#include <math.h>

typedef __attribute__((ext_vector_type(16))) _Float16 v16h;
typedef __attribute__((ext_vector_type(8)))  _Float16 v8h;
typedef __attribute__((ext_vector_type(8)))  float    v8f;

#define PAD_TOK 30
#define B_ 4
#define T_ 8192
#define V_ 38
#define E_ 100
#define EP 128      // E padded to 4 K-chunks of 32
#define H_ 256
#define O_ 38
#define DEC 12

// CDNA5 async global->LDS copy (ASYNCcnt-tracked), ISA §15.18.3 opcode 97.
// ldsaddr = 32-bit LDS byte address (generic shared addr low 32 bits per aperture rule).
__device__ __forceinline__ void async_copy_b64(unsigned int ldsaddr, const void* gsrc) {
  unsigned long long ga = (unsigned long long)(uintptr_t)gsrc;
  asm volatile("global_load_async_to_lds_b64 %0, %1, off"
               :: "v"(ldsaddr), "v"(ga) : "memory");
}
__device__ __forceinline__ void wait_async0() {
  asm volatile("s_wait_asynccnt 0x0" ::: "memory");
}

// ---------------- kernel 0: convert / pad weights to f16 ----------------
__global__ void k_convert(const float* __restrict__ emb, const float* __restrict__ wih,
                          const float* __restrict__ whh,
                          _Float16* __restrict__ emb_h, _Float16* __restrict__ wih_h,
                          _Float16* __restrict__ whh_h) {
  int i = blockIdx.x * blockDim.x + threadIdx.x;
  if (i < V_ * EP) {
    int v = i / EP, e = i % EP;
    emb_h[i] = (e < E_) ? (_Float16)emb[v * E_ + e] : (_Float16)0.f;
  }
  if (i < H_ * EP) {
    int n = i / EP, e = i % EP;
    wih_h[i] = (e < E_) ? (_Float16)wih[n * E_ + e] : (_Float16)0.f;
  }
  if (i < H_ * H_) whh_h[i] = (_Float16)whh[i];
}

// ---------------- kernel 1: sequence lengths (x != PAD) ----------------
__global__ void k_lengths(const long long* __restrict__ x, int* __restrict__ lengths) {
  __shared__ int red[256];
  int b = blockIdx.x;
  int cnt = 0;
  for (int t = threadIdx.x; t < T_; t += blockDim.x)
    cnt += (x[(long long)b * T_ + t] != (long long)PAD_TOK) ? 1 : 0;
  red[threadIdx.x] = cnt;
  __syncthreads();
  for (int s = 128; s > 0; s >>= 1) {
    if ((int)threadIdx.x < s) red[threadIdx.x] += red[threadIdx.x + s];
    __syncthreads();
  }
  if (threadIdx.x == 0) lengths[b] = red[0];
}

// ------- kernel 2: pre[t][b][n] = emb[x[b,t]]@W_ih^T + b_ih + b_hh (WMMA) -------
__global__ void k_precompute(const long long* __restrict__ x, const _Float16* __restrict__ emb_h,
                             const _Float16* __restrict__ wih_h, const float* __restrict__ b_ih,
                             const float* __restrict__ b_hh, float* __restrict__ pre) {
  const int b    = blockIdx.x & 3;
  const int t0   = (blockIdx.x >> 2) * 16;
  const int lane = threadIdx.x & 31;
  const int wave = threadIdx.x >> 5;   // 0..15, owns 16 output columns
  const int hl   = lane & 15;
  const int hi   = lane >> 4;          // 0 = lanes 0-15, 1 = lanes 16-31
  const int n    = wave * 16 + hl;     // output column (H)
  const int m    = hl;                 // A row = t offset within tile

  const int tok = (int)x[(long long)b * T_ + t0 + m];
  const _Float16* arow = emb_h + tok * EP;
  const _Float16* brow = wih_h + (long long)n * EP;

  v8f c0 = {}, c1 = {};                // two independent accumulator chains
#pragma unroll
  for (int kc = 0; kc < 4; ++kc) {
    const int ka = kc * 32 + hi * 16;  // A 16x32 f16: 16 contiguous K per lane-half
    v8h a_lo = *(const v8h*)(arow + ka);
    v8h a_hi = *(const v8h*)(arow + ka + 8);
    v16h a = __builtin_shufflevector(a_lo, a_hi, 0,1,2,3,4,5,6,7,8,9,10,11,12,13,14,15);
    const int kb = kc * 32 + hi * 16;  // B 32x16 f16: 16 contiguous K per lane-half
    v16h bf = *(const v16h*)(brow + kb);
    if (kc & 1) c1 = __builtin_amdgcn_wmma_f32_16x16x32_f16(false, a, false, bf, (short)0, c1, false, false);
    else        c0 = __builtin_amdgcn_wmma_f32_16x16x32_f16(false, a, false, bf, (short)0, c0, false, false);
  }
  const float bias = b_ih[n] + b_hh[n];
#pragma unroll
  for (int r = 0; r < 8; ++r) {
    const int M = r + hi * 8;          // C layout: VGPR r -> row r / r+8 by lane half
    pre[(long long)(t0 + M) * (B_ * H_) + b * H_ + n] = c0[r] + c1[r] + bias;
  }
}

// ------- kernel 3: serial recurrence (W_hh resident in WMMA B-fragments) + decode -------
__global__ void __launch_bounds__(512)
k_rnn(const float* __restrict__ emb_f, const float* __restrict__ wih_f,
      const float* __restrict__ b_ih, const float* __restrict__ b_hh,
      const float* __restrict__ wfc, const float* __restrict__ b_fc,
      const _Float16* __restrict__ whh_h, const float* __restrict__ pre,
      const int* __restrict__ lengths, float* __restrict__ out) {
  __shared__ _Float16 hbuf[16][264];                 // hidden state, rows 4..15 stay zero
  __shared__ __align__(16) float preLDS[2][B_ * H_]; // double-buffered pre[t] slab (2x4KB)
  __shared__ float    preIn[B_][H_];
  __shared__ float    logits[B_][O_];
  __shared__ int      toks[B_];
  __shared__ int      lenS[B_];

  const int tid  = threadIdx.x;
  const int lane = tid & 31;
  const int wave = tid >> 5;
  const int hl   = lane & 15;
  const int hi   = lane >> 4;
  const int n    = wave * 16 + hl;       // this lane's output column

  // Persistent B-fragments of W_hh^T: B[k][n] = W_hh[n][k] (row-major, contiguous k)
  v16h bfrag[8];
#pragma unroll
  for (int kc = 0; kc < 8; ++kc)
    bfrag[kc] = *(const v16h*)(whh_h + (long long)n * H_ + kc * 32 + hi * 16);

  if (tid < B_) lenS[tid] = lengths[tid];
  for (int i = tid; i < 16 * 264; i += blockDim.x) (&hbuf[0][0])[i] = (_Float16)0.f;

  // LDS byte addresses for this thread's 8-byte async slice in each buffer
  const unsigned int ldsA =
      (unsigned int)(uintptr_t)(&preLDS[0][0]) + (unsigned int)tid * 8u;
  const unsigned int ldsB =
      (unsigned int)(uintptr_t)(&preLDS[1][0]) + (unsigned int)tid * 8u;

  // Prologue: stream pre[0] into buffer 0
  async_copy_b64(ldsA, pre + 0 * (B_ * H_) + tid * 2);
  wait_async0();
  __syncthreads();

  float hkeep[8];
#pragma unroll
  for (int r = 0; r < 8; ++r) hkeep[r] = 0.f;

  // ---------------- encoder scan: 8192 sequential steps ----------------
  for (int t = 0; t < T_; ++t) {
    if (t + 16 < T_) {  // keep the stream warm in L2 well ahead (global_prefetch_b8)
      __builtin_prefetch(pre + (long long)(t + 16) * (B_ * H_) + tid, 0, 0);
      __builtin_prefetch(pre + (long long)(t + 16) * (B_ * H_) + 512 + tid, 0, 0);
    }
    // C-init from the LDS-resident pre[t] slab
    const float* pcur = &preLDS[t & 1][0];
    v8f c0 = {}, c1 = {};
#pragma unroll
    for (int r = 0; r < 4; ++r)
      if (hi == 0) c0[r] = pcur[r * H_ + n];
    // Overlap: async-stream pre[t+1] into the other buffer while we compute
    if (t + 1 < T_)
      async_copy_b64((t & 1) ? ldsA : ldsB, pre + (long long)(t + 1) * (B_ * H_) + tid * 2);
#pragma unroll
    for (int kc = 0; kc < 8; ++kc) {
      const int ka = kc * 32 + hi * 16;
      v8h a_lo = *(const v8h*)&hbuf[hl][ka];
      v8h a_hi = *(const v8h*)&hbuf[hl][ka + 8];
      v16h a = __builtin_shufflevector(a_lo, a_hi, 0,1,2,3,4,5,6,7,8,9,10,11,12,13,14,15);
      if (kc & 1) c1 = __builtin_amdgcn_wmma_f32_16x16x32_f16(false, a, false, bfrag[kc], (short)0, c1, false, false);
      else        c0 = __builtin_amdgcn_wmma_f32_16x16x32_f16(false, a, false, bfrag[kc], (short)0, c0, false, false);
    }
    __syncthreads();  // all waves done reading h(t) and preLDS[t&1]
#pragma unroll
    for (int r = 0; r < 8; ++r) {
      const int M = r + hi * 8;
      float hv;
      if (M < B_) hv = (t < lenS[M]) ? tanhf(c0[r] + c1[r]) : hkeep[r];  // pack_padded freeze
      else        hv = 0.f;
      hkeep[r] = hv;
      hbuf[M][n] = (_Float16)hv;
    }
    wait_async0();    // this wave's pre[t+1] slice has landed in LDS
    __syncthreads();  // h(t+1) + preLDS[(t+1)&1] visible to all waves
  }

  // ---------------- tok0 = argmax(out_last @ W_fc^T + b_fc) ----------------
  if (tid < B_ * O_) {
    int m = tid / O_, o = tid % O_;
    float s = b_fc[o];
    if (lenS[m] == T_) {  // pad_packed: zeros unless full-length
      float acc = 0.f;
      for (int k = 0; k < H_; ++k) acc += (float)hbuf[m][k] * wfc[o * H_ + k];
      s += acc;
    }
    logits[m][o] = s;
  }
  __syncthreads();
  if (tid < B_) {
    int best = 0; float bv = logits[tid][0];
    for (int o = 1; o < O_; ++o) if (logits[tid][o] > bv) { bv = logits[tid][o]; best = o; }
    toks[tid] = best;
    out[tid * (DEC + 1) + 0] = (float)best;
  }
  __syncthreads();

  // ---------------- 12-step autoregressive decode ----------------
  for (int d = 0; d < DEC; ++d) {
    for (int p = tid; p < B_ * H_; p += blockDim.x) {   // e @ W_ih^T + biases (scalar, tiny)
      int m = p >> 8, nn = p & 255;
      const float* er = emb_f + toks[m] * E_;
      const float* wr = wih_f + nn * E_;
      float acc = b_ih[nn] + b_hh[nn];
      for (int e = 0; e < E_; ++e) acc += er[e] * wr[e];
      preIn[m][nn] = acc;
    }
    __syncthreads();
    v8f c0 = {}, c1 = {};
#pragma unroll
    for (int r = 0; r < 4; ++r)
      if (hi == 0) c0[r] = preIn[r][n];
#pragma unroll
    for (int kc = 0; kc < 8; ++kc) {
      const int ka = kc * 32 + hi * 16;
      v8h a_lo = *(const v8h*)&hbuf[hl][ka];
      v8h a_hi = *(const v8h*)&hbuf[hl][ka + 8];
      v16h a = __builtin_shufflevector(a_lo, a_hi, 0,1,2,3,4,5,6,7,8,9,10,11,12,13,14,15);
      if (kc & 1) c1 = __builtin_amdgcn_wmma_f32_16x16x32_f16(false, a, false, bfrag[kc], (short)0, c1, false, false);
      else        c0 = __builtin_amdgcn_wmma_f32_16x16x32_f16(false, a, false, bfrag[kc], (short)0, c0, false, false);
    }
    __syncthreads();
#pragma unroll
    for (int r = 0; r < 8; ++r) {
      const int M = r + hi * 8;
      float hv = (M < B_) ? tanhf(c0[r] + c1[r]) : 0.f;
      hbuf[M][n] = (_Float16)hv;
    }
    __syncthreads();
    if (tid < B_ * O_) {
      int m = tid / O_, o = tid % O_;
      float acc = b_fc[o];
      for (int k = 0; k < H_; ++k) acc += (float)hbuf[m][k] * wfc[o * H_ + k];
      logits[m][o] = acc;
    }
    __syncthreads();
    if (tid < B_) {
      int best = 0; float bv = logits[tid][0];
      for (int o = 1; o < O_; ++o) if (logits[tid][o] > bv) { bv = logits[tid][o]; best = o; }
      toks[tid] = best;
      out[tid * (DEC + 1) + 1 + d] = (float)best;
    }
    __syncthreads();
  }
}

// ---------------------------------------------------------------------------
extern "C" void kernel_launch(void* const* d_in, const int* in_sizes, int n_in,
                              void* d_out, int out_size, void* d_ws, size_t ws_size,
                              hipStream_t stream) {
  const long long* x   = (const long long*)d_in[0];  // int64 tokens [4][8192]
  const float*     emb = (const float*)d_in[1];      // [38][100]
  const float*     wih = (const float*)d_in[2];      // [256][100]
  const float*     whh = (const float*)d_in[3];      // [256][256]
  const float*     bih = (const float*)d_in[4];      // [256]
  const float*     bhh = (const float*)d_in[5];      // [256]
  const float*     wfc = (const float*)d_in[6];      // [38][256]
  const float*     bfc = (const float*)d_in[7];      // [38]
  float*           out = (float*)d_out;              // [4][13]

  char*  ws  = (char*)d_ws;
  size_t off = 0;
  auto alloc = [&](size_t bytes) -> char* {
    off = (off + 255) & ~(size_t)255;
    char* p = ws + off;
    off += bytes;
    return p;
  };
  int*      lengths = (int*)     alloc(B_ * sizeof(int));
  _Float16* emb_h   = (_Float16*)alloc((size_t)V_ * EP * 2);
  _Float16* wih_h   = (_Float16*)alloc((size_t)H_ * EP * 2);
  _Float16* whh_h   = (_Float16*)alloc((size_t)H_ * H_ * 2);
  float*    pre     = (float*)   alloc((size_t)T_ * B_ * H_ * 4);  // 32 MB

  k_convert   <<<256, 256, 0, stream>>>(emb, wih, whh, emb_h, wih_h, whh_h);
  k_lengths   <<<B_, 256, 0, stream>>>(x, lengths);
  k_precompute<<<B_ * (T_ / 16), 512, 0, stream>>>(x, emb_h, wih_h, bih, bhh, pre);
  k_rnn       <<<1, 512, 0, stream>>>(emb, wih, bih, bhh, wfc, bfc, whh_h, pre, lengths, out);
}
